// Chooser_32229434589352
// MI455X (gfx1250) — matrix operand order repaired and checked by reading, not compiled, not run on hardware
//
#include <hip/hip_runtime.h>
#include <hip/hip_bf16.h>
#include <math.h>

typedef __attribute__((ext_vector_type(16))) _Float16 v16h;
typedef __attribute__((ext_vector_type(8)))  _Float16 v8h;
typedef __attribute__((ext_vector_type(8)))  float    v8f;
typedef __attribute__((ext_vector_type(4)))  float    v4f;

// ---- fast, branch-free transcendentals on CDNA5 hardware units ----
__device__ __forceinline__ float fast_exp2(float x) {
#if __has_builtin(__builtin_amdgcn_exp2f)
    return __builtin_amdgcn_exp2f(x);   // v_exp_f32
#else
    return exp2f(x);
#endif
}

__device__ __forceinline__ float fast_rcp(float x) {
#if __has_builtin(__builtin_amdgcn_rcpf)
    return __builtin_amdgcn_rcpf(x);    // v_rcp_f32
#else
    return 1.0f / x;
#endif
}

__device__ __forceinline__ float fast_tanh(float x) {
#if __has_builtin(__builtin_amdgcn_tanhf)
    return __builtin_amdgcn_tanhf(x);   // CDNA5 v_tanh_f32 (single TRANS op)
#else
    float ax = fabsf(x);
    float e2 = fast_exp2(ax * 2.8853900817779268f);  // exp(2*ax)
    float r  = 1.0f - 2.0f * fast_rcp(e2 + 1.0f);
    return copysignf(r, x);
#endif
}

// exp(x) for x <= 0 (post max-subtraction): plain v_exp_f32, underflow -> 0
__device__ __forceinline__ float fast_exp_nonpos(float x) {
    return fast_exp2(x * 1.4426950408889634f);
}

// ---- monotonic float<->uint encoding for atomicMax over signed floats ----
__device__ __forceinline__ unsigned fenc(float f) {
    unsigned u = __float_as_uint(f);
    return (u & 0x80000000u) ? ~u : (u | 0x80000000u);
}
__device__ __forceinline__ float fdec(unsigned u) {
    u = (u & 0x80000000u) ? (u & 0x7FFFFFFFu) : ~u;
    return __uint_as_float(u);
}

// ---- WMMA fragment helpers (CDNA5 16x16x32 f16 layouts) ----
// A (16x32 f16): lane L: row m = L&15; slots 0-7 hold K=k0+kg+0..7,
// slots 8-15 hold K=k0+kg+16..23 where kg = (L>>4)*8.
__device__ __forceinline__ v16h load_a_global(const float* xrow, int k0, int lane) {
    int kg = ((lane >> 4) & 1) * 8;
    const float* p0 = xrow + k0 + kg;        // slots 0-7
    const float* p1 = xrow + k0 + kg + 16;   // slots 8-15
    v4f f0 = *(const v4f*)(p0);
    v4f f1 = *(const v4f*)(p0 + 4);
    v4f f2 = *(const v4f*)(p1);
    v4f f3 = *(const v4f*)(p1 + 4);
    v16h a;
#pragma unroll
    for (int i = 0; i < 4; ++i) {
        a[i]      = (_Float16)f0[i];
        a[4 + i]  = (_Float16)f1[i];
        a[8 + i]  = (_Float16)f2[i];
        a[12 + i] = (_Float16)f3[i];
    }
    return a;
}

__device__ __forceinline__ v16h load_a_lds(const _Float16* base, int stride, int k0, int lane) {
    int m  = lane & 15;
    int kg = ((lane >> 4) & 1) * 8;
    const _Float16* row = base + m * stride + k0 + kg;
    v8h lo = *(const v8h*)(row);        // K = k0+kg+0..7
    v8h hi = *(const v8h*)(row + 16);   // K = k0+kg+16..23
    v16h a;
#pragma unroll
    for (int i = 0; i < 8; ++i) { a[i] = lo[i]; a[8 + i] = hi[i]; }
    return a;
}

// B (32x16 f16): lane L holds column n=L&15, rows K=(L>>4)*8 + (j&7) + (j>>3)*16 (j=0..15).
// prep kernel stores tiles pre-swizzled so each lane does ONE 32B load.
__device__ __forceinline__ v16h load_b(const _Float16* tile, int lane) {
    return *(const v16h*)(tile + lane * 16);
}

// D (16x16 f32): lane L holds column n=L&15; VGPR r holds row m=r+(L>>4)*8.
__device__ __forceinline__ void store_d_lds(_Float16* base, int stride, int n0,
                                            const v8f& c, float bias, int lane) {
    int n  = n0 + (lane & 15);
    int mg = ((lane >> 4) & 1) * 8;
#pragma unroll
    for (int r = 0; r < 8; ++r)
        base[(mg + r) * stride + n] = (_Float16)fast_tanh(c[r] + bias);
}

__device__ __forceinline__ v8f wmma16(const v16h& a, const v16h& b, const v8f& c) {
    return __builtin_amdgcn_wmma_f32_16x16x32_f16(false, a, false, b, (short)0, c, false, false);
}

// =====================================================================
// Kernel 1: weight repack to f16 fragments + segment offset scan + tail copy
// ws tile order: W1 (kt*8+nt, 16 tiles), W2 (kt*4+nt, 16 tiles), W3 (2 tiles)
// =====================================================================
__global__ void prep_kernel(const float* __restrict__ W1, const float* __restrict__ W2,
                            const float* __restrict__ W3, const int* __restrict__ sizes,
                            int nseg, unsigned* gmax, int* offsets, _Float16* frag,
                            int* out_tail, int tail_count) {
    int tid = threadIdx.x;
    if (tid == 0) {
        *gmax = 0u;  // below fenc of every float: safe identity for max
        int acc = 0;
        for (int i = 0; i < nseg; ++i) { offsets[i] = acc; acc += sizes[i]; }
        offsets[nseg] = acc;
    }
    for (int i = tid; i < tail_count; i += blockDim.x) out_tail[i] = sizes[i];

    const int NELEM = 34 * 512;
    for (int idx = tid; idx < NELEM; idx += blockDim.x) {
        int tile = idx >> 9;
        int e    = idx & 511;
        int lane = e >> 4;
        int j    = e & 15;
        const float* W; int Ncols, k0, n0;
        if (tile < 16)      { W = W1; Ncols = 128; k0 = (tile >> 3) * 32; n0 = (tile & 7) * 16; }
        else if (tile < 32) { int r = tile - 16; W = W2; Ncols = 64; k0 = (r >> 2) * 32; n0 = (r & 3) * 16; }
        else                { W = W3; Ncols = 16; k0 = (tile - 32) * 32; n0 = 0; }
        int k = k0 + ((lane >> 4) & 1) * 8 + (j & 7) + (j >> 3) * 16;
        int n = n0 + (lane & 15);
        frag[idx] = (_Float16)W[k * Ncols + n];
    }
}

// =====================================================================
// Kernel 2: WMMA MLP -> logits, block max -> atomic global max
// 128 threads = 4 waves; each wave owns TWO 16-token tiles (32 tokens),
// so each B fragment load is amortized over 2 WMMAs (halves L0 B-traffic).
// =====================================================================
__global__ void __launch_bounds__(128)
mlp_kernel(const float* __restrict__ x, const float* __restrict__ b1,
           const float* __restrict__ b2, const float* __restrict__ b3,
           const float* __restrict__ W4, const float* __restrict__ b4,
           const _Float16* __restrict__ frag, unsigned* gmax,
           float* __restrict__ logits, int total) {
    __shared__ __align__(16) _Float16 hbuf[4][2][16 * 128];  // 32 KB
    __shared__ float red[128];

    int tid  = threadIdx.x;
    int lane = tid & 31;
    int wave = tid >> 5;
    long t0  = ((long)blockIdx.x * 4 + wave) * 32;
    if (t0 + 32 > total) t0 = (long)total - 32;  // clamp tail: duplicates write identical values

    const _Float16* w1f = frag;
    const _Float16* w2f = frag + 16 * 512;
    const _Float16* w3f = frag + 32 * 512;
    int m = lane & 15;

    // -------- Layer 1: 2 x [16x64] x [64x128], 32 WMMA --------
    v16h a0[2], a1[2];
#pragma unroll
    for (int t = 0; t < 2; ++t) {
        const float* xrow = x + (size_t)(t0 + t * 16 + m) * 64;
        a0[t] = load_a_global(xrow, 0, lane);
        a1[t] = load_a_global(xrow, 32, lane);
    }
#pragma unroll
    for (int n = 0; n < 8; ++n) {
        v16h bk0 = load_b(w1f + (0 * 8 + n) * 512, lane);
        v16h bk1 = load_b(w1f + (1 * 8 + n) * 512, lane);
        float bias = b1[n * 16 + m];
#pragma unroll
        for (int t = 0; t < 2; ++t) {
            v8f c = {0.f, 0.f, 0.f, 0.f, 0.f, 0.f, 0.f, 0.f};
            c = wmma16(a0[t], bk0, c);
            c = wmma16(a1[t], bk1, c);
            store_d_lds(hbuf[wave][t], 128, n * 16, c, bias, lane);
        }
    }
    __syncthreads();

    // -------- Layer 2: 2 x [16x128] x [128x64], 32 WMMA --------
    v16h a2[2][4];
#pragma unroll
    for (int t = 0; t < 2; ++t)
#pragma unroll
        for (int kt = 0; kt < 4; ++kt)
            a2[t][kt] = load_a_lds(hbuf[wave][t], 128, kt * 32, lane);
    __syncthreads();
#pragma unroll
    for (int n = 0; n < 4; ++n) {
        v16h bk[4];
#pragma unroll
        for (int kt = 0; kt < 4; ++kt) bk[kt] = load_b(w2f + (kt * 4 + n) * 512, lane);
        float bias = b2[n * 16 + m];
#pragma unroll
        for (int t = 0; t < 2; ++t) {
            v8f c = {0.f, 0.f, 0.f, 0.f, 0.f, 0.f, 0.f, 0.f};
#pragma unroll
            for (int kt = 0; kt < 4; ++kt) c = wmma16(a2[t][kt], bk[kt], c);
            store_d_lds(hbuf[wave][t], 64, n * 16, c, bias, lane);
        }
    }
    __syncthreads();

    // -------- Layer 3: 2 x [16x64] x [64x16], 4 WMMA --------
    v16h a3[2][2];
#pragma unroll
    for (int t = 0; t < 2; ++t)
#pragma unroll
        for (int kt = 0; kt < 2; ++kt)
            a3[t][kt] = load_a_lds(hbuf[wave][t], 64, kt * 32, lane);
    __syncthreads();
    {
        v16h bk0 = load_b(w3f, lane);
        v16h bk1 = load_b(w3f + 512, lane);
        float bias = b3[m];
#pragma unroll
        for (int t = 0; t < 2; ++t) {
            v8f c = {0.f, 0.f, 0.f, 0.f, 0.f, 0.f, 0.f, 0.f};
            c = wmma16(a3[t][0], bk0, c);
            c = wmma16(a3[t][1], bk1, c);
            store_d_lds(hbuf[wave][t], 16, 0, c, bias, lane);
        }
    }
    __syncthreads();

    // -------- Layer 4: [16x16] x [16x1] per-row dot; all 32 lanes active --------
    // lane 0..15 -> tile 0 rows 0..15; lane 16..31 -> tile 1 rows 0..15
    {
        const _Float16* row = hbuf[wave][lane >> 4] + (lane & 15) * 16;
        float acc = b4[0];
#pragma unroll
        for (int j = 0; j < 16; ++j) acc += (float)row[j] * W4[j];
        logits[t0 + lane] = acc;
        red[tid] = acc;
    }
    __syncthreads();
#pragma unroll
    for (int s = 64; s > 0; s >>= 1) {
        if (tid < s) red[tid] = fmaxf(red[tid], red[tid + s]);
        __syncthreads();
    }
    if (tid == 0) atomicMax(gmax, fenc(red[0]));
}

// =====================================================================
// Kernel 3: per-segment softmax in place over logits held in d_out
// =====================================================================
__global__ void __launch_bounds__(256)
softmax_kernel(const int* __restrict__ offsets, const unsigned* __restrict__ gmax,
               float* __restrict__ out) {
    __shared__ float red[256];
    __shared__ float sdenom;
    int b   = blockIdx.x;
    int off = offsets[b];
    int end = offsets[b + 1];
    float g = fdec(*gmax);

    float s = 0.f;
    for (int i = off + threadIdx.x; i < end; i += blockDim.x) {
        float e = fast_exp_nonpos(out[i] - g);   // arg <= 0, v_exp_f32 is safe
        out[i] = e;
        s += e;
    }
    red[threadIdx.x] = s;
    __syncthreads();
#pragma unroll
    for (int st = 128; st > 0; st >>= 1) {
        if (threadIdx.x < st) red[threadIdx.x] += red[threadIdx.x + st];
        __syncthreads();
    }
    if (threadIdx.x == 0) sdenom = red[0];
    __syncthreads();
    float inv = 1.0f / sdenom;
    for (int i = off + threadIdx.x; i < end; i += blockDim.x) out[i] *= inv;
}

// =====================================================================
extern "C" void kernel_launch(void* const* d_in, const int* in_sizes, int n_in,
                              void* d_out, int out_size, void* d_ws, size_t ws_size,
                              hipStream_t stream) {
    const float* x     = (const float*)d_in[0];
    const int*   sizes = (const int*)d_in[1];
    const float* W1    = (const float*)d_in[2];
    const float* b1    = (const float*)d_in[3];
    const float* W2    = (const float*)d_in[4];
    const float* b2    = (const float*)d_in[5];
    const float* W3    = (const float*)d_in[6];
    const float* b3    = (const float*)d_in[7];
    const float* W4    = (const float*)d_in[8];
    const float* b4    = (const float*)d_in[9];

    int total = in_sizes[0] / 64;   // tokens
    int nseg  = in_sizes[1];        // segments
    float* out = (float*)d_out;

    // workspace layout: [gmax u32][pad][offsets int[nseg+1]][pad to 32][frags f16 34*512]
    char* ws = (char*)d_ws;
    unsigned* gmax  = (unsigned*)ws;
    int* offsets    = (int*)(ws + 16);
    size_t fragOff  = (16 + sizeof(int) * (size_t)(nseg + 1) + 31) & ~(size_t)31;
    _Float16* frag  = (_Float16*)(ws + fragOff);

    // tuple return (out, sizes): sizes raw int32 bits appended after TOTAL floats
    int tail_count = out_size - total;
    if (tail_count < 0) tail_count = 0;
    if (tail_count > nseg) tail_count = nseg;
    int* out_tail = (int*)d_out + total;

    prep_kernel<<<1, 256, 0, stream>>>(W1, W2, W3, sizes, nseg, gmax, offsets, frag,
                                       out_tail, tail_count);

    // 4 waves/block, 32 tokens/wave -> 128 tokens per block
    long nwaves  = ((long)total + 31) / 32;
    int  nblocks = (int)((nwaves + 3) / 4);
    mlp_kernel<<<nblocks, 128, 0, stream>>>(x, b1, b2, b3, W4, b4, frag, gmax, out, total);

    softmax_kernel<<<nseg, 256, 0, stream>>>(offsets, gmax, out);
}
